// HOGAT_51616916963827
// MI455X (gfx1250) — matrix operand order repaired and checked by reading, not compiled, not run on hardware
//
#include <hip/hip_runtime.h>
#include <hip/hip_bf16.h>
#include <stdint.h>

// ---------------- problem constants ----------------
#define N_NODES  10000
#define N_MOTIFS 2500
#define N_AUG    12500
#define N_EDGES  500000
#define E_TOT    (N_EDGES + N_AUG)   // edges + self loops = 512500
#define NFEAT    512
#define NHID     128
#define OUT1     (2*NHID)            // 256
#define M_PAD    12512               // N_AUG padded to multiple of 16
#define M_TILES  (M_PAD/16)          // 782
#define ADJ_WAVES 4
#define ADJ_NT    2                  // n-tiles per wave in adj kernel

typedef __attribute__((ext_vector_type(2)))  float  v2f;
typedef __attribute__((ext_vector_type(8)))  float  v8f;
typedef __attribute__((ext_vector_type(16))) __bf16 v16bf;

union Frag32B { uint4 u[2]; v16bf v; };

// ---------------- helpers ----------------
__device__ inline unsigned ordF(float x) {
    unsigned u = __float_as_uint(x);
    return (u & 0x80000000u) ? ~u : (u | 0x80000000u);
}
__device__ inline float unordF(unsigned u) {
    return (u & 0x80000000u) ? __uint_as_float(u & 0x7FFFFFFFu)
                             : __uint_as_float(~u);
}
__device__ inline void edge_sd(const int* __restrict__ eidx, int i, int& s, int& d) {
    if (i < N_EDGES) { s = eidx[i]; d = eidx[N_EDGES + i]; }
    else             { s = d = i - N_EDGES; }   // self loops
}
// fast sigmoid: v_mul + v_exp + v_add + v_rcp (no IEEE div sequence)
__device__ inline float dsigmoid(float x) {
    return __builtin_amdgcn_rcpf(1.0f + __expf(-x));
}

// ---------------- x_aug build ----------------
__global__ void build_xaug_kernel(const float* __restrict__ x,
                                  const int* __restrict__ motif,
                                  float* __restrict__ xaug) {
    long long idx = (long long)blockIdx.x * blockDim.x + threadIdx.x;
    if (idx >= (long long)M_PAD * NFEAT) return;
    int row = (int)(idx / NFEAT), f = (int)(idx % NFEAT);
    float v = 0.0f;
    if (row < N_NODES) {
        v = x[(size_t)row * NFEAT + f];
    } else if (row < N_AUG) {
        int mi = row - N_NODES;
        float s = 0.0f;
        #pragma unroll
        for (int j = 0; j < 4; ++j)
            s += x[(size_t)motif[mi * 4 + j] * NFEAT + f];
        v = 0.25f * s;
    }
    xaug[(size_t)row * NFEAT + f] = v;
}

// ---------------- weight transpose: in[K][N] -> out[N][K] ----------------
__global__ void transpose_kernel(const float* __restrict__ in, float* __restrict__ out,
                                 int K, int N) {
    long long idx = (long long)blockIdx.x * blockDim.x + threadIdx.x;
    if (idx >= (long long)K * N) return;
    int k = (int)(idx / N), n = (int)(idx % N);
    out[(size_t)n * K + k] = in[idx];
}

// ---------------- f32 WMMA GEMM, B pre-transposed [N][K] ----------------
// 4 waves per block, one 16x16 tile each along N. Inner loop:
//   global_load_b64 (A frag) + global_load_b64 (B frag) + v_wmma_f32_16x16x4_f32
template<bool SIG>
__global__ void gemm_f32_bt_kernel(const float* __restrict__ A, const float* __restrict__ Bt,
                                   float* __restrict__ C, const float* __restrict__ bias,
                                   int M_valid, int N, int K) {
    const int wave = threadIdx.x >> 5, lane = threadIdx.x & 31;
    const int half = lane >> 4, r16 = lane & 15;
    const int tn = blockIdx.x * 4 + wave;
    if (tn * 16 >= N) return;                  // whole-wave exit
    const int m0 = blockIdx.y * 16, n0 = tn * 16;

    const float* __restrict__ Arow = A  + (size_t)(m0 + r16) * K + 2 * half;
    const float* __restrict__ Brow = Bt + (size_t)(n0 + r16) * K + 2 * half;

    v8f c = {};
    #pragma unroll 4
    for (int k = 0; k < K; k += 4) {
        v2f a = *reinterpret_cast<const v2f*>(Arow + k);
        v2f b = *reinterpret_cast<const v2f*>(Brow + k);
        c = __builtin_amdgcn_wmma_f32_16x16x4_f32(false, a, false, b,
                                                  (short)0, c, false, false);
    }
    #pragma unroll
    for (int r = 0; r < 8; ++r) {
        int m = m0 + r + 8 * half;
        if (m < M_valid) {
            float v = c[r];
            if (SIG) v = dsigmoid(v + bias[n0 + r16]);
            C[(size_t)m * N + n0 + r16] = v;
        }
    }
}

// ---------------- GAT attention kernels ----------------
__global__ void init_gat_kernel(float* __restrict__ out, const float* __restrict__ bias,
                                int OUTF, unsigned* __restrict__ m_ord,
                                float* __restrict__ denom) {
    long long idx = (long long)blockIdx.x * blockDim.x + threadIdx.x;
    if (idx < (long long)M_PAD * OUTF) out[idx] = bias[idx % OUTF];
    if (idx < M_PAD) { m_ord[idx] = 0u; denom[idx] = 0.0f; }
}

__global__ void alpha_kernel(const float* __restrict__ h,
                             const float* __restrict__ a_s, const float* __restrict__ a_d,
                             float* __restrict__ as_out, float* __restrict__ ad_out,
                             int OUTF) {
    int n = blockIdx.x * blockDim.x + threadIdx.x;
    if (n >= M_PAD) return;
    float ss = 0.0f, sd = 0.0f;
    const float* hp = h + (size_t)n * OUTF;
    for (int f = 0; f < OUTF; ++f) {
        float hv = hp[f];
        ss = fmaf(hv, a_s[f], ss);
        sd = fmaf(hv, a_d[f], sd);
    }
    as_out[n] = ss; ad_out[n] = sd;
}

__global__ void edge_max_kernel(const int* __restrict__ eidx,
                                const float* __restrict__ a_s, const float* __restrict__ a_d,
                                float* __restrict__ e_buf, unsigned* __restrict__ m_ord) {
    int i = blockIdx.x * blockDim.x + threadIdx.x;
    if (i >= E_TOT) return;
    int s, d; edge_sd(eidx, i, s, d);
    float e = a_s[s] + a_d[d];
    e = (e > 0.0f) ? e : 0.2f * e;          // leaky_relu(0.2)
    e_buf[i] = e;
    atomicMax(&m_ord[d], ordF(e));
}

__global__ void edge_w_kernel(const int* __restrict__ eidx,
                              float* __restrict__ e_buf,
                              const unsigned* __restrict__ m_ord,
                              float* __restrict__ denom) {
    int i = blockIdx.x * blockDim.x + threadIdx.x;
    if (i >= E_TOT) return;
    int s, d; edge_sd(eidx, i, s, d);
    float w = __expf(e_buf[i] - unordF(m_ord[d]));
    e_buf[i] = w;
    atomicAdd(&denom[d], w);
}

// one block per edge; blockDim.x == OUTF feature lanes
__global__ void edge_scatter_kernel(const int* __restrict__ eidx,
                                    const float* __restrict__ w_buf,
                                    const float* __restrict__ denom,
                                    const float* __restrict__ h,
                                    float* __restrict__ out) {
    __shared__ int ss, sd;
    __shared__ float scoef;
    int i = blockIdx.x;
    if (threadIdx.x == 0) {
        int s, d; edge_sd(eidx, i, s, d);
        ss = s; sd = d;
        scoef = w_buf[i] / (denom[d] + 1e-16f);
    }
    __syncthreads();
    int f = threadIdx.x;
    atomicAdd(&out[(size_t)sd * blockDim.x + f],
              scoef * h[(size_t)ss * blockDim.x + f]);
}

// ---------------- f32 -> bf16 pack of h2 ----------------
__global__ void f2bf_kernel(const float* __restrict__ src, __bf16* __restrict__ dst,
                            long long count) {
    long long idx = (long long)blockIdx.x * blockDim.x + threadIdx.x;
    if (idx < count) dst[idx] = (__bf16)src[idx];
}

// ---------------- rec_adj = sigmoid(h2 @ h2^T), bf16 WMMA 16x16x32 ----------------
// 4 waves per block; each wave computes TWO 16x16 n-tiles sharing one LDS-staged
// 16x128 A tile. A frags preloaded to registers -> 8 back-to-back WMMAs per wave.
__global__ void adj_wmma_bf16_kernel(const __bf16* __restrict__ H,
                                     float* __restrict__ out) {
    __shared__ __attribute__((aligned(16))) __bf16 As[16 * NHID];  // 4 KB
    const int t = threadIdx.x;                  // 0..127
    const int m0 = blockIdx.y * 16;
    {   // cooperative stage of A tile: 16 bf16 (2 x b128) per thread
        int row = t >> 3;              // 0..15
        int cseg = (t & 7) * 16;       // 0..112 step 16
        const uint4* g = reinterpret_cast<const uint4*>(H + (size_t)(m0 + row) * NHID + cseg);
        uint4* s = reinterpret_cast<uint4*>(&As[row * NHID + cseg]);
        s[0] = g[0]; s[1] = g[1];
    }
    __syncthreads();

    const int wave = t >> 5, lane = t & 31;
    const int half = lane >> 4, r16 = lane & 15;
    const int tnbase = blockIdx.x * (ADJ_WAVES * ADJ_NT);
    const int tn0 = tnbase + wave;
    const int tn1 = tnbase + ADJ_WAVES + wave;
    if (tn0 >= M_TILES) return;                  // whole-wave exit (after barrier)
    const bool v1 = (tn1 < M_TILES);
    const int tn1c = v1 ? tn1 : tn0;             // clamp: redundant compute, guarded store

    // preload all 4 A fragments from LDS (2 x ds_load_b128 each)
    Frag32B fa[4];
    #pragma unroll
    for (int kk = 0; kk < 4; ++kk) {
        const __bf16* ap = &As[r16 * NHID + kk * 32 + 16 * half];
        fa[kk].u[0] = *reinterpret_cast<const uint4*>(ap);
        fa[kk].u[1] = *reinterpret_cast<const uint4*>(ap + 8);
    }

    const __bf16* __restrict__ B0 = H + (size_t)(tn0  * 16 + r16) * NHID + 16 * half;
    const __bf16* __restrict__ B1 = H + (size_t)(tn1c * 16 + r16) * NHID + 16 * half;

    v8f c0 = {}, c1 = {};
    #pragma unroll
    for (int kk = 0; kk < 4; ++kk) {
        Frag32B fb0, fb1;
        const uint4* bp0 = reinterpret_cast<const uint4*>(B0 + kk * 32);
        const uint4* bp1 = reinterpret_cast<const uint4*>(B1 + kk * 32);
        fb0.u[0] = bp0[0]; fb0.u[1] = bp0[1];
        fb1.u[0] = bp1[0]; fb1.u[1] = bp1[1];
        c0 = __builtin_amdgcn_wmma_f32_16x16x32_bf16(false, fa[kk].v, false, fb0.v,
                                                     (short)0, c0, false, false);
        c1 = __builtin_amdgcn_wmma_f32_16x16x32_bf16(false, fa[kk].v, false, fb1.v,
                                                     (short)0, c1, false, false);
    }
    #pragma unroll
    for (int r = 0; r < 8; ++r) {
        int m = m0 + r + 8 * half;
        int n0 = tn0 * 16 + r16;
        if (m < N_AUG && n0 < N_AUG)
            out[(size_t)m * N_AUG + n0] = dsigmoid(c0[r]);
        if (v1) {
            int n1 = tn1 * 16 + r16;
            if (m < N_AUG && n1 < N_AUG)
                out[(size_t)m * N_AUG + n1] = dsigmoid(c1[r]);
        }
    }
}

// ---------------- host-side GAT layer driver ----------------
static void run_gat_layer(const float* h, float* out,
                          const float* a_s, const float* a_d, const float* bias, int OUTF,
                          const int* eidx, float* e_buf,
                          float* alpha_s, float* alpha_d,
                          unsigned* m_ord, float* denom, hipStream_t stream) {
    long long tot = (long long)M_PAD * OUTF;
    init_gat_kernel<<<(unsigned)((tot + 255) / 256), 256, 0, stream>>>(out, bias, OUTF, m_ord, denom);
    alpha_kernel<<<(M_PAD + 255) / 256, 256, 0, stream>>>(h, a_s, a_d, alpha_s, alpha_d, OUTF);
    edge_max_kernel<<<(E_TOT + 255) / 256, 256, 0, stream>>>(eidx, alpha_s, alpha_d, e_buf, m_ord);
    edge_w_kernel<<<(E_TOT + 255) / 256, 256, 0, stream>>>(eidx, e_buf, m_ord, denom);
    edge_scatter_kernel<<<E_TOT, OUTF, 0, stream>>>(eidx, e_buf, denom, h, out);
}

extern "C" void kernel_launch(void* const* d_in, const int* in_sizes, int n_in,
                              void* d_out, int out_size, void* d_ws, size_t ws_size,
                              hipStream_t stream) {
    (void)in_sizes; (void)n_in; (void)out_size; (void)ws_size;
    const float* x     = (const float*)d_in[0];
    const int*   eidx  = (const int*)  d_in[1];
    const int*   motif = (const int*)  d_in[2];
    const float* W1    = (const float*)d_in[3];
    const float* a_s1  = (const float*)d_in[4];
    const float* a_d1  = (const float*)d_in[5];
    const float* b1    = (const float*)d_in[6];
    const float* W2    = (const float*)d_in[7];
    const float* a_s2  = (const float*)d_in[8];
    const float* a_d2  = (const float*)d_in[9];
    const float* b2    = (const float*)d_in[10];
    const float* W_lin = (const float*)d_in[11];
    const float* b_lin = (const float*)d_in[12];

    // workspace layout (all 16B-aligned)
    char* ws = (char*)d_ws;
    float*    bufA   = (float*)ws;                                    // M_PAD*NFEAT (x_aug, later g1_out)
    float*    bufB   = (float*)(ws += (size_t)M_PAD * NFEAT * 4, ws); // M_PAD*OUT1 (g1_h, later g2_h)
    float*    bufC   = (float*)(ws += (size_t)M_PAD * OUT1 * 4, ws);  // M_PAD*NHID (h2)
    __bf16*   h2bf   = (__bf16*)(ws += (size_t)M_PAD * NHID * 4, ws); // M_PAD*NHID bf16
    float*    e_buf  = (float*)(ws += (size_t)M_PAD * NHID * 2, ws);  // E_TOT
    float*    alphaS = (float*)(ws += (size_t)E_TOT * 4, ws);
    float*    alphaD = (float*)(ws += (size_t)M_PAD * 4, ws);
    unsigned* m_ord  = (unsigned*)(ws += (size_t)M_PAD * 4, ws);
    float*    denom  = (float*)(ws += (size_t)M_PAD * 4, ws);
    float*    W1t    = (float*)(ws += (size_t)M_PAD * 4, ws);         // [OUT1][NFEAT]
    float*    W2t    = (float*)(ws += (size_t)NFEAT * OUT1 * 4, ws);  // [NHID][OUT1]

    float* rec_feat = (float*)d_out;                        // [N_AUG x NFEAT]
    float* rec_adj  = rec_feat + (size_t)N_AUG * NFEAT;     // [N_AUG x N_AUG]

    // 0. transpose weights once (W_lin is already [N][K] for its GEMM)
    transpose_kernel<<<(NFEAT * OUT1 + 255) / 256, 256, 0, stream>>>(W1, W1t, NFEAT, OUT1);
    transpose_kernel<<<(OUT1 * NHID + 255) / 256, 256, 0, stream>>>(W2, W2t, OUT1, NHID);

    // 1. build augmented features
    build_xaug_kernel<<<(unsigned)(((long long)M_PAD * NFEAT + 255) / 256), 256, 0, stream>>>(
        x, motif, bufA);

    // 2. g1_h = x_aug @ W1  [M_PAD x 256], K=512
    gemm_f32_bt_kernel<false><<<dim3(OUT1 / 16 / 4, M_TILES), 128, 0, stream>>>(
        bufA, W1t, bufB, nullptr, M_PAD, OUT1, NFEAT);

    // 3. GAT layer 1 -> g1_out (reuses x_aug buffer)
    float* g1_out = bufA;
    run_gat_layer(bufB, g1_out, a_s1, a_d1, b1, OUT1,
                  eidx, e_buf, alphaS, alphaD, m_ord, denom, stream);

    // 4. g2_h = g1_out @ W2  [M_PAD x 128], K=256 (overwrites g1_h)
    gemm_f32_bt_kernel<false><<<dim3(NHID / 16 / 4, M_TILES), 128, 0, stream>>>(
        g1_out, W2t, bufB, nullptr, M_PAD, NHID, OUT1);

    // 5. GAT layer 2 -> h2
    run_gat_layer(bufB, bufC, a_s2, a_d2, b2, NHID,
                  eidx, e_buf, alphaS, alphaD, m_ord, denom, stream);

    // 6. rec_feature = sigmoid(h2 @ W_lin^T + b_lin)  [12500 x 512], K=128
    gemm_f32_bt_kernel<true><<<dim3(NFEAT / 16 / 4, M_TILES), 128, 0, stream>>>(
        bufC, W_lin, rec_feat, b_lin, N_AUG, NFEAT, NHID);

    // 7. rec_adj = sigmoid(h2 @ h2^T) via bf16 WMMA
    f2bf_kernel<<<(unsigned)(((long long)M_PAD * NHID + 255) / 256), 256, 0, stream>>>(
        bufC, h2bf, (long long)M_PAD * NHID);
    adj_wmma_bf16_kernel<<<dim3((M_TILES + ADJ_WAVES * ADJ_NT - 1) / (ADJ_WAVES * ADJ_NT),
                                M_TILES),
                           32 * ADJ_WAVES, 0, stream>>>(h2bf, rec_adj);
}
